// MultiHeadAttention_53815940219243
// MI455X (gfx1250) — matrix operand order repaired and checked by reading, not compiled, not run on hardware
//
#include <hip/hip_runtime.h>
#include <hip/hip_bf16.h>
#include <math.h>

// ---------------------------------------------------------------------------
// MI455X (gfx1250) attention pipeline, f16 WMMA, all hot GEMMs in NT form,
// LDS double-buffered. Staging uses the CDNA5 async global->LDS engine
// (GLOBAL_LOAD_ASYNC_TO_LDS_B128 / ASYNCcnt) when the toolchain exposes it,
// falling back to global_load+ds_store otherwise.
//   q = x@Wq + bq ; k = x@Wk + bk          (NT GEMM vs pre-transposed weights)
//   Wvo^T = Wo^T@Wv ; cvo = bv@Wo ; vo = x@Wvo + cvo     (folded v-path)
//   S = q@k^T (f16 in ws) ; per-row max & sum(exp)
//   out = exp(S - m)/(l*8) @ vo + bo       (f32 out)
// ---------------------------------------------------------------------------

typedef __attribute__((ext_vector_type(16))) _Float16 v16h;
typedef __attribute__((ext_vector_type(8)))  _Float16 v8h;
typedef __attribute__((ext_vector_type(8)))  float    v8f;
typedef __attribute__((ext_vector_type(4)))  int      v4i;

#define SEQ   4096
#define DMOD  1024
#define HID   64

#if defined(__has_builtin)
#if __has_builtin(__builtin_amdgcn_global_load_async_to_lds_b128) && \
    __has_builtin(__builtin_amdgcn_s_wait_asynccnt)
#define HAVE_ASYNC_LDS 1
#endif
#endif
#ifndef HAVE_ASYNC_LDS
#define HAVE_ASYNC_LDS 0
#endif

typedef __attribute__((address_space(1))) v4i* gv4i_ptr;
typedef __attribute__((address_space(3))) v4i* lv4i_ptr;

// 16B global -> LDS copy chunk (async engine when available).
__device__ __forceinline__ void copy16_g2l(_Float16* ldst, const _Float16* gsrc) {
#if HAVE_ASYNC_LDS
  __builtin_amdgcn_global_load_async_to_lds_b128((gv4i_ptr)gsrc, (lv4i_ptr)ldst,
                                                 0, 0);
#else
  *(v8h*)ldst = *(const v8h*)gsrc;
#endif
}
__device__ __forceinline__ void wait_async() {
#if HAVE_ASYNC_LDS
  __builtin_amdgcn_s_wait_asynccnt(0);
#endif
}

// ---- fragment loaders from LDS tiles (stride 32 halves) -------------------
// A 16x32 tile, row-major [row][k]: lane l holds row l%16,
// K = (j%8) + (j/8)*16 + (l/16)*8  ->  two contiguous 16B chunks.
__device__ __forceinline__ v16h lds_load_a(const _Float16* base) {
  int lane = threadIdx.x & 31;
  const _Float16* p = base + (lane & 15) * 32 + (lane >> 4) * 8;
  v8h lo = *(const v8h*)p;
  v8h hi = *(const v8h*)(p + 16);
  return __builtin_shufflevector(lo, hi, 0,1,2,3,4,5,6,7,8,9,10,11,12,13,14,15);
}
// B 32x16 tile stored K-contiguous per column [col][k]: lane l holds col l%16,
// K = (j%8) + (j/8)*16 + (l/16)*8  ->  two contiguous 16B chunks.
__device__ __forceinline__ v16h lds_load_b(const _Float16* base) {
  int lane = threadIdx.x & 31;
  const _Float16* p = base + (lane & 15) * 32 + (lane >> 4) * 8;
  v8h lo = *(const v8h*)p;
  v8h hi = *(const v8h*)(p + 16);
  return __builtin_shufflevector(lo, hi, 0,1,2,3,4,5,6,7,8,9,10,11,12,13,14,15);
}

__device__ __forceinline__ v8f wmma_f16(v16h a, v16h b, v8f c) {
  return __builtin_amdgcn_wmma_f32_16x16x32_f16(false, a, false, b,
                                                (short)0, c, false, false);
}

// ---- f32 -> f16 convert ---------------------------------------------------
__global__ void cvt_kernel(const float* __restrict__ in,
                           _Float16* __restrict__ out, int n) {
  int i = blockIdx.x * blockDim.x + threadIdx.x;
  if (i < n) out[i] = (_Float16)in[i];
}

// ---- f32 [R x C] -> f16 transposed [C x R] --------------------------------
__global__ void cvt_t_kernel(const float* __restrict__ in,
                             _Float16* __restrict__ out, int R, int C) {
  int i = blockIdx.x * blockDim.x + threadIdx.x;
  if (i < R * C) {
    int r = i / C, c = i % C;
    out[(size_t)c * R + r] = (_Float16)in[i];
  }
}

// ---- cvo[64] = bv @ Wo ----------------------------------------------------
__global__ void biasvo_kernel(const float* __restrict__ bv,
                              const float* __restrict__ Wo,
                              float* __restrict__ cvo) {
  int j = threadIdx.x;  // 64 threads
  float s = 0.f;
  for (int kk = 0; kk < DMOD; ++kk) s += bv[kk] * Wo[kk * HID + j];
  cvo[j] = s;
}

// ---------------------------------------------------------------------------
// C[MxN] f16 = A[MxK] @ Bk[NxK]^T (+bias[n]).  Block: 128 threads (4 wave32),
// 128x64 tile; each wave 32x64 (2 A-frags x 4 B-frags = 8 WMMA / K-step).
// LDS double-buffered; staging via async global->LDS when available.
// ---------------------------------------------------------------------------
__global__ void gemm_nt_kernel(const _Float16* __restrict__ A,
                               const _Float16* __restrict__ Bk,
                               const float* __restrict__ bias,
                               _Float16* __restrict__ C,
                               int M, int N, int K) {
  __shared__ __align__(16) _Float16 As[2][128 * 32];
  __shared__ __align__(16) _Float16 Bs[2][64 * 32];
  int wid = threadIdx.x >> 5, lane = threadIdx.x & 31;
  size_t tile_m = (size_t)blockIdx.x * 128, tile_n = (size_t)blockIdx.y * 64;
  v8f acc[2][4] = {};

  auto stage = [&](int buf, int k0) {
#pragma unroll
    for (int it = 0; it < 4; ++it) {  // A: 512 x 8h chunks
      int idx = threadIdx.x + it * 128;
      int row = idx >> 2, c8 = idx & 3;
      copy16_g2l(As[buf] + row * 32 + c8 * 8,
                 A + (tile_m + row) * K + k0 + c8 * 8);
    }
#pragma unroll
    for (int it = 0; it < 2; ++it) {  // B: 256 x 8h chunks
      int idx = threadIdx.x + it * 128;
      int row = idx >> 2, c8 = idx & 3;
      copy16_g2l(Bs[buf] + row * 32 + c8 * 8,
                 Bk + (tile_n + row) * K + k0 + c8 * 8);
    }
  };

  int T = K / 32;
  stage(0, 0);
  wait_async();
  __syncthreads();
  for (int kt = 0; kt < T; ++kt) {
    int cur = kt & 1;
    if (kt + 1 < T) stage(cur ^ 1, (kt + 1) * 32);  // runs under WMMA below
    v16h a0 = lds_load_a(As[cur] + (wid * 32) * 32);
    v16h a1 = lds_load_a(As[cur] + (wid * 32 + 16) * 32);
#pragma unroll
    for (int t = 0; t < 4; ++t) {
      v16h b = lds_load_b(Bs[cur] + t * 16 * 32);
      acc[0][t] = wmma_f16(a0, b, acc[0][t]);
      acc[1][t] = wmma_f16(a1, b, acc[1][t]);
    }
    wait_async();
    __syncthreads();
  }

  int col0 = lane & 15, half = lane >> 4;
#pragma unroll
  for (int s = 0; s < 2; ++s)
#pragma unroll
    for (int t = 0; t < 4; ++t)
#pragma unroll
      for (int r = 0; r < 8; ++r) {
        size_t m = tile_m + wid * 32 + s * 16 + r + half * 8;
        size_t n = tile_n + t * 16 + col0;
        float v = acc[s][t][r];
        if (bias) v += bias[n];
        C[m * N + n] = (_Float16)v;
      }
}

// ---- per-row max / sum(exp) ----------------------------------------------
__device__ __forceinline__ float wave_max(float v) {
  for (int m = 16; m; m >>= 1) v = fmaxf(v, __shfl_xor(v, m));
  return v;
}
__device__ __forceinline__ float wave_sum(float v) {
  for (int m = 16; m; m >>= 1) v += __shfl_xor(v, m);
  return v;
}
__global__ void rowstats_kernel(const _Float16* __restrict__ S,
                                float* __restrict__ rmax,
                                float* __restrict__ rsum, int N) {
  int row = blockIdx.x;
  __shared__ __align__(16) _Float16 buf[SEQ];
  __shared__ float red[8];
  __shared__ float statm;
  const _Float16* src = S + (size_t)row * N;
  for (int i = threadIdx.x; i < N / 8; i += blockDim.x)
    copy16_g2l(buf + i * 8, src + i * 8);
  wait_async();
  __syncthreads();
  float mx = -3.0e38f;
  for (int j = threadIdx.x; j < N; j += blockDim.x)
    mx = fmaxf(mx, (float)buf[j]);
  mx = wave_max(mx);
  int wid = threadIdx.x >> 5, lane = threadIdx.x & 31;
  if (lane == 0) red[wid] = mx;
  __syncthreads();
  if (threadIdx.x == 0) {
    float m = red[0];
    for (int i = 1; i < 8; ++i) m = fmaxf(m, red[i]);
    statm = m;
  }
  __syncthreads();
  float bm = statm;
  float s = 0.f;
  for (int j = threadIdx.x; j < N; j += blockDim.x)
    s += __expf((float)buf[j] - bm);
  s = wave_sum(s);
  __syncthreads();
  if (lane == 0) red[wid] = s;
  __syncthreads();
  if (threadIdx.x == 0) {
    float t = 0.f;
    for (int i = 0; i < 8; ++i) t += red[i];
    rmax[row] = bm;
    rsum[row] = t;
  }
}

// ---- out[Mx64] f32 = exp(S - m)/(l*8) @ VO + bo ---------------------------
__global__ void pv_kernel(const _Float16* __restrict__ S,
                          const float* __restrict__ rmax,
                          const float* __restrict__ rsum,
                          const _Float16* __restrict__ VO,  // [K x 64] row-major
                          const float* __restrict__ bo,
                          float* __restrict__ Out, int M, int K) {
  __shared__ __align__(16) _Float16 As[64 * 32];
  __shared__ __align__(16) _Float16 Bts[64 * 32];
  int wid = threadIdx.x >> 5, lane = threadIdx.x & 31;
  int tile_m = blockIdx.x * 64;
  v8f acc[4] = {};
  for (int k0 = 0; k0 < K; k0 += 32) {
    __syncthreads();
    for (int it = 0; it < 16; ++it) {    // exp() applied during staging
      int idx = threadIdx.x + it * 128;  // 0..2047
      int row = idx >> 5, kk = idx & 31;
      float s = (float)S[(size_t)(tile_m + row) * K + k0 + kk];
      As[row * 32 + kk] = (_Float16)__expf(s - rmax[tile_m + row]);
    }
    for (int it = 0; it < 16; ++it) {
      int idx = threadIdx.x + it * 128;
      int kk = idx >> 6, col = idx & 63;
      Bts[col * 32 + kk] = VO[(size_t)(k0 + kk) * HID + col];
    }
    __syncthreads();
    v16h a = lds_load_a(As + wid * 16 * 32);
#pragma unroll
    for (int t = 0; t < 4; ++t) {
      v16h b = lds_load_b(Bts + t * 16 * 32);
      acc[t] = wmma_f16(a, b, acc[t]);
    }
  }
  int col0 = lane & 15, half = lane >> 4;
#pragma unroll
  for (int t = 0; t < 4; ++t)
#pragma unroll
    for (int r = 0; r < 8; ++r) {
      int m = tile_m + wid * 16 + r + half * 8;
      int n = t * 16 + col0;
      Out[(size_t)m * HID + n] = acc[t][r] / (rsum[m] * 8.0f) + bo[n];
    }
}

// ---------------------------------------------------------------------------
extern "C" void kernel_launch(void* const* d_in, const int* in_sizes, int n_in,
                              void* d_out, int out_size, void* d_ws,
                              size_t ws_size, hipStream_t stream) {
  (void)in_sizes; (void)n_in; (void)out_size; (void)ws_size;
  const float* x  = (const float*)d_in[0];
  const float* Wq = (const float*)d_in[1];
  const float* bq = (const float*)d_in[2];
  const float* Wk = (const float*)d_in[3];
  const float* bk = (const float*)d_in[4];
  const float* Wv = (const float*)d_in[5];
  const float* bv = (const float*)d_in[6];
  const float* Wo = (const float*)d_in[7];
  const float* bo = (const float*)d_in[8];
  float* out = (float*)d_out;

  char* ws = (char*)d_ws;
  size_t off = 0;
  auto alloc = [&](size_t bytes) -> char* {
    char* p = ws + off;
    off = (off + bytes + 255) & ~(size_t)255;
    return p;
  };
  _Float16* xh    = (_Float16*)alloc((size_t)SEQ * DMOD * 2);
  _Float16* wqth  = (_Float16*)alloc((size_t)DMOD * DMOD * 2);   // Wq^T
  _Float16* wkth  = (_Float16*)alloc((size_t)DMOD * DMOD * 2);   // Wk^T
  _Float16* wvh   = (_Float16*)alloc((size_t)DMOD * DMOD * 2);   // Wv (plain)
  _Float16* woth  = (_Float16*)alloc((size_t)128 * DMOD * 2);    // Wo^T (64 rows, padded)
  _Float16* wvoth = (_Float16*)alloc((size_t)128 * DMOD * 2);    // Wvo^T (padded)
  _Float16* qh    = (_Float16*)alloc((size_t)SEQ * DMOD * 2);
  _Float16* kh    = (_Float16*)alloc((size_t)SEQ * DMOD * 2);
  _Float16* voh   = (_Float16*)alloc((size_t)SEQ * HID * 2);
  float*    cvo   = (float*)alloc(HID * 4);
  _Float16* sc    = (_Float16*)alloc((size_t)SEQ * SEQ * 2);     // 32 MB
  float*    rm    = (float*)alloc(SEQ * 4);
  float*    rl    = (float*)alloc(SEQ * 4);

  // 1) converts (weights transposed so all hot GEMMs are NT form)
  cvt_kernel<<<(SEQ * DMOD) / 256, 256, 0, stream>>>(x, xh, SEQ * DMOD);
  cvt_t_kernel<<<(DMOD * DMOD) / 256, 256, 0, stream>>>(Wq, wqth, DMOD, DMOD);
  cvt_t_kernel<<<(DMOD * DMOD) / 256, 256, 0, stream>>>(Wk, wkth, DMOD, DMOD);
  cvt_kernel<<<(DMOD * DMOD) / 256, 256, 0, stream>>>(Wv, wvh, DMOD * DMOD);
  cvt_t_kernel<<<(DMOD * HID) / 256, 256, 0, stream>>>(Wo, woth, DMOD, HID);

  // 2) folded v-path: Wvo^T[j][i] = sum_k Wo^T[j][k] * Wv[i][k] ; cvo = bv@Wo
  biasvo_kernel<<<1, HID, 0, stream>>>(bv, Wo, cvo);
  gemm_nt_kernel<<<dim3(1, DMOD / 64), 128, 0, stream>>>(
      woth, wvh, nullptr, wvoth, 128, DMOD, DMOD);

  // 3) projections: C[m][n] = sum_k A[m][k] * Bk[n][k]
  gemm_nt_kernel<<<dim3(SEQ / 128, DMOD / 64), 128, 0, stream>>>(
      xh, wqth, bq, qh, SEQ, DMOD, DMOD);
  gemm_nt_kernel<<<dim3(SEQ / 128, DMOD / 64), 128, 0, stream>>>(
      xh, wkth, bk, kh, SEQ, DMOD, DMOD);
  gemm_nt_kernel<<<dim3(SEQ / 128, 1), 128, 0, stream>>>(
      xh, wvoth, cvo, voh, SEQ, HID, DMOD);

  // 4) scores = q @ k^T
  gemm_nt_kernel<<<dim3(SEQ / 128, SEQ / 64), 128, 0, stream>>>(
      qh, kh, nullptr, sc, SEQ, SEQ, DMOD);

  // 5) softmax stats
  rowstats_kernel<<<SEQ, 256, 0, stream>>>(sc, rm, rl, SEQ);

  // 6) out = softmax(S)/8 @ vo + bo
  pv_kernel<<<dim3(SEQ / 64, 1), 128, 0, stream>>>(sc, rm, rl, voh, bo, out,
                                                   SEQ, SEQ);
}
